// ResidualGATModel_88527865905193
// MI455X (gfx1250) — matrix operand order repaired and checked by reading, not compiled, not run on hardware
//
#include <hip/hip_runtime.h>
#include <hip/hip_bf16.h>
#include <stdint.h>

// ---------------------------------------------------------------------------
// ResidualGATModel forward for MI455X (gfx1250, wave32, WMMA).
// 256x256 GEMMs on v_wmma_f32_16x16x32_f16, 64x256 block tiles, 4 waves,
// 4x4 accumulators/wave, double-buffered fragments. The edge-path GEMM fuses:
//   prologue: previous layer's  e += relu(BN(e_tmp))  (read e+e_tmp, write e)
//   epilogue: +bu +Vx_i +Vx_j, e_tmp store, sigmoid gate-mean (LDS reduce),
//             BN channel stats (LDS partials + global f32 atomics)
//   head mode: relu(.+bm) @ Wout fused reduction -> out (h never hits memory)
// This halves traffic on the memory-bound 67MB edge tensor (23.3 TB/s HBM).
// ---------------------------------------------------------------------------

typedef __attribute__((ext_vector_type(16))) _Float16 v16h;
typedef __attribute__((ext_vector_type(8)))  float    v8f;

#define Bsz 4
#define Vn  128
#define Hn  256
#define NHn 8
#define HDn 32
#define Ln  4
#define EROWS (Bsz * Vn * Vn)   // 65536
#define XROWS (Bsz * Vn)        // 512

// ---------------------------------------------------------------------------
__global__ void f32_to_f16(const float* __restrict__ src, _Float16* __restrict__ dst, int n) {
  int i = blockIdx.x * blockDim.x + threadIdx.x;
  if (i < n) dst[i] = (_Float16)src[i];
}

__global__ void zero_buf(float* __restrict__ p, int n) {
  int i = blockIdx.x * blockDim.x + threadIdx.x;
  if (i < n) p[i] = 0.0f;
}

// ---------------------------------------------------------------------------
// Fragment loader: 4 A-frags (LDS tile) + 4 B-frags (f16 weights, global).
// A 16x32 f16 frag: lane m = lane&15, K halves split by lane>>4.
// B 32x16 f16 frag: lane col N = lane&15, K halves split by lane>>4 (same K order as A).
// ---------------------------------------------------------------------------
__device__ __forceinline__ void load_frags(v16h (&af)[4], v16h (&bf)[4],
                                           const _Float16* __restrict__ As,
                                           const uint32_t* __restrict__ W32,
                                           int K0, int mloc, int hi, int lane, int n0) {
#pragma unroll
  for (int mt = 0; mt < 4; ++mt) {
#pragma unroll
    for (int v = 0; v < 8; ++v) {
      int k = K0 + ((v < 4) ? (hi * 8 + 2 * v) : (16 + hi * 8 + 2 * (v - 4)));
      af[mt][2 * v]     = As[(mt * 16 + mloc) * 256 + k];
      af[mt][2 * v + 1] = As[(mt * 16 + mloc) * 256 + k + 1];
    }
  }
  const _Float16* Wh = (const _Float16*)W32;
  const int nb = n0 + (lane & 15);
#pragma unroll
  for (int nt = 0; nt < 4; ++nt) {
#pragma unroll
    for (int v = 0; v < 8; ++v) {
      int k = K0 + ((v < 4) ? (hi * 8 + 2 * v) : (16 + hi * 8 + 2 * (v - 4)));
      bf[nt][2 * v]     = Wh[k * 256 + nb + nt * 16];
      bf[nt][2 * v + 1] = Wh[(k + 1) * 256 + nb + nt * 16];
    }
  }
}

// ---------------------------------------------------------------------------
// Generic WMMA GEMM (node path): C[M,256] = A[M,256] @ W16 + bias
// grid.x = M/64, block = 128 threads (4 waves).
// ---------------------------------------------------------------------------
__global__ __launch_bounds__(128, 1)
void gemm256_wmma(const float* __restrict__ A, const _Float16* __restrict__ W16,
                  const float* __restrict__ bias, float* __restrict__ C, int relu) {
  __shared__ _Float16 As[64 * 256];  // 32 KB
  const int tid = threadIdx.x;
  const long row0 = (long)blockIdx.x * 64;

  for (int i = tid; i < 64 * 256; i += 128)
    As[i] = (_Float16)A[row0 * 256 + i];
  __syncthreads();

  const int wave = tid >> 5, lane = tid & 31;
  const int n0 = wave * 64, mloc = lane & 15, hi = lane >> 4;

  v8f acc[4][4];
#pragma unroll
  for (int mt = 0; mt < 4; ++mt)
#pragma unroll
    for (int nt = 0; nt < 4; ++nt) acc[mt][nt] = (v8f){};

  const uint32_t* W32 = (const uint32_t*)W16;
  v16h af0[4], bf0[4], af1[4], bf1[4];
  load_frags(af0, bf0, As, W32, 0, mloc, hi, lane, n0);

#pragma unroll
  for (int ks = 0; ks < 8; ++ks) {
    v16h(&afc)[4] = (ks & 1) ? af1 : af0;
    v16h(&bfc)[4] = (ks & 1) ? bf1 : bf0;
    v16h(&afn)[4] = (ks & 1) ? af0 : af1;
    v16h(&bfn)[4] = (ks & 1) ? bf0 : bf1;
    if (ks < 7) load_frags(afn, bfn, As, W32, (ks + 1) * 32, mloc, hi, lane, n0);
#pragma unroll
    for (int mt = 0; mt < 4; ++mt)
#pragma unroll
      for (int nt = 0; nt < 4; ++nt)
        acc[mt][nt] = __builtin_amdgcn_wmma_f32_16x16x32_f16(
            false, afc[mt], false, bfc[nt], (short)0, acc[mt][nt], false, false);
  }

#pragma unroll
  for (int mt = 0; mt < 4; ++mt)
#pragma unroll
    for (int nt = 0; nt < 4; ++nt) {
      const int ncol = n0 + nt * 16 + (lane & 15);
      const float bb = bias[ncol];
#pragma unroll
      for (int i = 0; i < 8; ++i) {
        const long r = row0 + mt * 16 + i + hi * 8;
        float vv = acc[mt][nt][i] + bb;
        if (relu) vv = vv > 0.f ? vv : 0.f;
        C[r * 256 + ncol] = vv;
      }
    }
}

// ---------------------------------------------------------------------------
// Edge-path fused WMMA GEMM. grid.x = EROWS/64, block = 128.
//  prologue : if EtmpPrev, e += relu(BN(EtmpPrev)) using (usum,usq,ug,ubb);
//             write back to Ewb (if non-null); convert tile to f16 LDS.
//  core     : A @ W16  (W = Wu[l] or Wm)
//  epilogue : mode 0 (layer): t = acc + bias + Vx_i + Vx_j; EtmpOut = t;
//             gmean[row] = mean_c sigmoid(t); atomics into (osum,osq).
//             mode 1 (head): h = relu(acc + bias); out[row,:] = h @ Wout + bout.
// ---------------------------------------------------------------------------
__global__ __launch_bounds__(128, 1)
void edge_gemm_fused(const float* __restrict__ E, float* __restrict__ Ewb,
                     const float* __restrict__ EtmpPrev,
                     const float* __restrict__ usum, const float* __restrict__ usq,
                     const float* __restrict__ ug, const float* __restrict__ ubb,
                     float uInvN,
                     const _Float16* __restrict__ W16, const float* __restrict__ bias,
                     int mode,
                     const float* __restrict__ Vx, float* __restrict__ EtmpOut,
                     float* __restrict__ gmean, float* __restrict__ osum,
                     float* __restrict__ osq,
                     const float* __restrict__ Wout, const float* __restrict__ bout,
                     float* __restrict__ out) {
  __shared__ _Float16 As[64 * 256];  // 32 KB
  __shared__ float gpart[64];
  __shared__ float chansum[256];
  __shared__ float chansq[256];

  const int tid = threadIdx.x;
  const long row0 = (long)blockIdx.x * 64;

  // ---- prologue: residual BN+relu update of previous layer (fused) ----
  for (int i = tid; i < 64 * 256; i += 128) {
    const long gi = row0 * 256 + i;
    float v = E[gi];
    if (EtmpPrev) {
      const int c = i & 255;
      const float mu  = usum[c] * uInvN;
      const float var = usq[c] * uInvN - mu * mu;
      float y = ug[c] * (EtmpPrev[gi] - mu) * rsqrtf(var + 1e-5f) + ubb[c];
      v += y > 0.f ? y : 0.f;
      if (Ewb) Ewb[gi] = v;
    }
    As[i] = (_Float16)v;
  }
  // zero epilogue LDS accumulators
  for (int i = tid; i < 64 + 256 + 256; i += 128) {
    if (i < 64) gpart[i] = 0.f;
    else if (i < 320) chansum[i - 64] = 0.f;
    else chansq[i - 320] = 0.f;
  }
  __syncthreads();

  const int wave = tid >> 5, lane = tid & 31;
  const int n0 = wave * 64, mloc = lane & 15, hi = lane >> 4;

  v8f acc[4][4];
#pragma unroll
  for (int mt = 0; mt < 4; ++mt)
#pragma unroll
    for (int nt = 0; nt < 4; ++nt) acc[mt][nt] = (v8f){};

  const uint32_t* W32 = (const uint32_t*)W16;
  v16h af0[4], bf0[4], af1[4], bf1[4];
  load_frags(af0, bf0, As, W32, 0, mloc, hi, lane, n0);

#pragma unroll
  for (int ks = 0; ks < 8; ++ks) {
    v16h(&afc)[4] = (ks & 1) ? af1 : af0;
    v16h(&bfc)[4] = (ks & 1) ? bf1 : bf0;
    v16h(&afn)[4] = (ks & 1) ? af0 : af1;
    v16h(&bfn)[4] = (ks & 1) ? bf0 : bf1;
    if (ks < 7) load_frags(afn, bfn, As, W32, (ks + 1) * 32, mloc, hi, lane, n0);
#pragma unroll
    for (int mt = 0; mt < 4; ++mt)
#pragma unroll
      for (int nt = 0; nt < 4; ++nt)
        acc[mt][nt] = __builtin_amdgcn_wmma_f32_16x16x32_f16(
            false, afc[mt], false, bfc[nt], (short)0, acc[mt][nt], false, false);
  }

  // block-constant row decomposition: rows row0..row0+63 share (b, i_node)
  const int bB    = (int)(row0 >> 14);
  const int inode = (int)((row0 >> 7) & 127);
  const int jbase = (int)(row0 & 127);           // 0 or 64

  // per-lane column constants (4 columns: one per nt)
  float bb4[4];
  int col4[4];
#pragma unroll
  for (int nt = 0; nt < 4; ++nt) {
    col4[nt] = n0 + nt * 16 + (lane & 15);
    bb4[nt] = bias[col4[nt]];
  }

  if (mode == 0) {
    float vxi4[4];
#pragma unroll
    for (int nt = 0; nt < 4; ++nt)
      vxi4[nt] = Vx[(long)(bB * 128 + inode) * 256 + col4[nt]];
    float cs[4] = {0.f, 0.f, 0.f, 0.f}, cq[4] = {0.f, 0.f, 0.f, 0.f};
#pragma unroll
    for (int mt = 0; mt < 4; ++mt) {
#pragma unroll
      for (int i = 0; i < 8; ++i) {
        const int lr = mt * 16 + i + hi * 8;     // 0..63
        const long r = row0 + lr;
        const int j = jbase + lr;                // no wrap: row0 % 64 == 0
        const long vxjBase = (long)(bB * 128 + j) * 256;
        float gp = 0.f;
#pragma unroll
        for (int nt = 0; nt < 4; ++nt) {
          const float t = acc[mt][nt][i] + bb4[nt] + vxi4[nt] + Vx[vxjBase + col4[nt]];
          EtmpOut[r * 256 + col4[nt]] = t;
          gp += 1.f / (1.f + __expf(-t));
          cs[nt] += t;
          cq[nt] += t * t;
        }
        atomicAdd(&gpart[lr], gp);
      }
    }
#pragma unroll
    for (int nt = 0; nt < 4; ++nt) {
      atomicAdd(&chansum[col4[nt]], cs[nt]);
      atomicAdd(&chansq[col4[nt]],  cq[nt]);
    }
    __syncthreads();
    if (tid < 64) gmean[row0 + tid] = gpart[tid] * (1.f / 256.f);
    for (int c = tid; c < 256; c += 128) {
      atomicAdd(&osum[c], chansum[c]);
      atomicAdd(&osq[c],  chansq[c]);
    }
  } else {
    // head: h = relu(acc + bm); out = h @ Wout + bout (Wout is 256x2)
    float w04[4], w14[4];
#pragma unroll
    for (int nt = 0; nt < 4; ++nt) {
      w04[nt] = Wout[col4[nt] * 2 + 0];
      w14[nt] = Wout[col4[nt] * 2 + 1];
    }
#pragma unroll
    for (int mt = 0; mt < 4; ++mt) {
#pragma unroll
      for (int i = 0; i < 8; ++i) {
        const int lr = mt * 16 + i + hi * 8;
        float o0 = 0.f, o1 = 0.f;
#pragma unroll
        for (int nt = 0; nt < 4; ++nt) {
          float h = acc[mt][nt][i] + bb4[nt];
          h = h > 0.f ? h : 0.f;
          o0 += h * w04[nt];
          o1 += h * w14[nt];
        }
        atomicAdd(&gpart[lr], o0);       // reuse gpart for out[:,0]
        atomicAdd(&chansum[lr], o1);     // reuse chansum[0..63] for out[:,1]
      }
    }
    __syncthreads();
    if (tid < 64) {
      out[(row0 + tid) * 2 + 0] = gpart[tid] + bout[0];
      out[(row0 + tid) * 2 + 1] = chansum[tid] + bout[1];
    }
  }
}

// ---------------------------------------------------------------------------
// GAT coordinate embedding, stage 1
// ---------------------------------------------------------------------------
__global__ __launch_bounds__(256)
void gat_xw(const float* __restrict__ coord, const float* __restrict__ Wg,
            const float* __restrict__ att_src, const float* __restrict__ att_dst,
            float* __restrict__ xw, float* __restrict__ a_s, float* __restrict__ a_d) {
  __shared__ float row[2048];
  __shared__ float red[256];
  const int bv = blockIdx.x;
  const int t  = threadIdx.x;
  const float c0 = coord[bv * 2 + 0], c1 = coord[bv * 2 + 1];
  for (int n = t; n < 2048; n += 256) {
    float v = c0 * Wg[n] + c1 * Wg[2048 + n];
    row[n] = v;
    xw[(long)bv * 2048 + n] = v;
  }
  __syncthreads();
  for (int h = 0; h < 8; ++h) {
    float ps = row[h * 256 + t] * att_src[h * 256 + t];
    float pd = row[h * 256 + t] * att_dst[h * 256 + t];
    red[t] = ps; __syncthreads();
    for (int off = 128; off > 0; off >>= 1) { if (t < off) red[t] += red[t + off]; __syncthreads(); }
    if (t == 0) a_s[bv * 8 + h] = red[0];
    __syncthreads();
    red[t] = pd; __syncthreads();
    for (int off = 128; off > 0; off >>= 1) { if (t < off) red[t] += red[t + off]; __syncthreads(); }
    if (t == 0) a_d[bv * 8 + h] = red[0];
    __syncthreads();
  }
}

// ---------------------------------------------------------------------------
// GAT stage 2: masked leaky-relu softmax over sources, head-mean aggregate
// ---------------------------------------------------------------------------
__global__ __launch_bounds__(256)
void gat_attn(const int* __restrict__ x_edges, const float* __restrict__ a_s,
              const float* __restrict__ a_d, const float* __restrict__ xw,
              const float* __restrict__ b_gat, float* __restrict__ x) {
  __shared__ float al[128 * 8];
  const int blk = blockIdx.x;
  const int b = blk >> 7, tt = blk & 127;
  const int tid = threadIdx.x;
  for (int p = tid; p < 1024; p += 256) {
    int s = p >> 3, h = p & 7;
    bool m = (x_edges[(b * 128 + s) * 128 + tt] != 0) || (s == tt);
    float z = a_d[(b * 128 + tt) * 8 + h] + a_s[(b * 128 + s) * 8 + h];
    z = z > 0.f ? z : 0.2f * z;
    al[s * 8 + h] = m ? z : -1e30f;
  }
  __syncthreads();
  if (tid < 8) {
    const int h = tid;
    float mx = -1e30f;
    for (int s = 0; s < 128; ++s) mx = fmaxf(mx, al[s * 8 + h]);
    float sum = 0.f;
    for (int s = 0; s < 128; ++s) { float p = __expf(al[s * 8 + h] - mx); al[s * 8 + h] = p; sum += p; }
    float inv = 1.f / sum;
    for (int s = 0; s < 128; ++s) al[s * 8 + h] *= inv;
  }
  __syncthreads();
  const int d = tid;
  float acc = 0.f;
  for (int h = 0; h < 8; ++h)
    for (int s = 0; s < 128; ++s)
      acc += al[s * 8 + h] * xw[(long)((b * 128 + s) * 8 + h) * 256 + d];
  x[(long)(b * 128 + tt) * 256 + d] = acc * 0.125f + b_gat[d];
}

// ---------------------------------------------------------------------------
// Edge embedding init
// ---------------------------------------------------------------------------
__global__ __launch_bounds__(256)
void e_init(const int* __restrict__ x_edges, const float* __restrict__ xev,
            const float* __restrict__ W_ev, const float* __restrict__ emb_edges,
            float* __restrict__ e) {
  long idx = (long)blockIdx.x * 256 + threadIdx.x;
  int c = (int)(idx & 255);
  long row = idx >> 8;
  float v;
  if (c < 128) v = xev[row] * W_ev[c];
  else         v = emb_edges[x_edges[row] * 128 + (c - 128)];
  e[idx] = v;
}

// ---------------------------------------------------------------------------
// Per-channel sum / sumsq over rows (node path only, 512 rows)
// ---------------------------------------------------------------------------
__global__ __launch_bounds__(256)
void chan_stats(const float* __restrict__ src, float* __restrict__ sum,
                float* __restrict__ sq, int rowsPerBlock) {
  const int c = threadIdx.x;
  const long r0 = (long)blockIdx.x * rowsPerBlock;
  float s = 0.f, s2 = 0.f;
  for (int r = 0; r < rowsPerBlock; ++r) {
    float v = src[(r0 + r) * 256 + c];
    s += v; s2 += v * v;
  }
  atomicAdd(&sum[c], s);
  atomicAdd(&sq[c], s2);
}

// ---------------------------------------------------------------------------
// Gated multi-head node attention: block per (b,h,i); 128 threads
// ---------------------------------------------------------------------------
__global__ __launch_bounds__(128)
void node_attention(const float* __restrict__ Q, const float* __restrict__ K,
                    const float* __restrict__ Vt, const float* __restrict__ gmean,
                    float* __restrict__ att) {
  __shared__ float sc[128];
  __shared__ float red[128];
  const int blk = blockIdx.x;
  const int b = blk >> 10, h = (blk >> 7) & 7, i = blk & 127;
  const int j = threadIdx.x;
  const float* qrow = Q + (long)(b * 128 + i) * 256 + h * 32;
  const float* krow = K + (long)(b * 128 + j) * 256 + h * 32;
  float d = 0.f;
  for (int k = 0; k < 32; ++k) d += qrow[k] * krow[k];
  d *= 0.17677669529663687f;                    // 1/sqrt(HD)
  d *= gmean[(long)(b * 128 + i) * 128 + j];
  sc[j] = d;
  red[j] = d; __syncthreads();
  for (int off = 64; off > 0; off >>= 1) { if (j < off) red[j] = fmaxf(red[j], red[j + off]); __syncthreads(); }
  const float mx = red[0];
  __syncthreads();
  const float p = __expf(sc[j] - mx);
  red[j] = p; __syncthreads();
  for (int off = 64; off > 0; off >>= 1) { if (j < off) red[j] += red[j + off]; __syncthreads(); }
  const float inv = 1.f / red[0];
  __syncthreads();
  sc[j] = p * inv;
  __syncthreads();
  if (j < 32) {
    float acc = 0.f;
    for (int s = 0; s < 128; ++s) acc += sc[s] * Vt[(long)(b * 128 + s) * 256 + h * 32 + j];
    att[(long)(b * 128 + i) * 256 + h * 32 + j] = acc;
  }
}

// ---------------------------------------------------------------------------
// dst += relu(batchnorm(tmp)) (node path only)
// ---------------------------------------------------------------------------
__global__ __launch_bounds__(256)
void residual_bn_relu(float* __restrict__ dst, const float* __restrict__ tmp,
                      const float* __restrict__ sum, const float* __restrict__ sq,
                      const float* __restrict__ g, const float* __restrict__ bb,
                      float invN) {
  const long idx = (long)blockIdx.x * 256 + threadIdx.x;
  const int c = (int)(idx & 255);
  const float mu  = sum[c] * invN;
  const float var = sq[c] * invN - mu * mu;
  float y = g[c] * (tmp[idx] - mu) * rsqrtf(var + 1e-5f) + bb[c];
  dst[idx] += y > 0.f ? y : 0.f;
}

// ---------------------------------------------------------------------------
// Host orchestration
// ---------------------------------------------------------------------------
extern "C" void kernel_launch(void* const* d_in, const int* in_sizes, int n_in,
                              void* d_out, int out_size, void* d_ws, size_t ws_size,
                              hipStream_t stream) {
  (void)in_sizes; (void)n_in; (void)out_size; (void)ws_size;

  const int*   x_edges        = (const int*)d_in[0];
  const float* x_edges_values = (const float*)d_in[1];
  /* d_in[2] x_nodes unused */
  const float* coord    = (const float*)d_in[3];
  const float* W_gat    = (const float*)d_in[4];
  const float* att_src  = (const float*)d_in[5];
  const float* att_dst  = (const float*)d_in[6];
  const float* b_gat    = (const float*)d_in[7];
  const float* W_ev     = (const float*)d_in[8];
  const float* emb_edges= (const float*)d_in[9];
  const float* Wu  = (const float*)d_in[10];
  const float* Wv  = (const float*)d_in[11];
  const float* Wq  = (const float*)d_in[12];
  const float* Wk  = (const float*)d_in[13];
  const float* Wval= (const float*)d_in[14];
  const float* Wo  = (const float*)d_in[15];
  const float* bu  = (const float*)d_in[16];
  const float* bv  = (const float*)d_in[17];
  const float* bq  = (const float*)d_in[18];
  const float* bk  = (const float*)d_in[19];
  const float* bval= (const float*)d_in[20];
  const float* bo  = (const float*)d_in[21];
  const float* g_node = (const float*)d_in[22];
  const float* be_node= (const float*)d_in[23];
  const float* g_edge = (const float*)d_in[24];
  const float* be_edge= (const float*)d_in[25];
  const float* Wm   = (const float*)d_in[26];
  const float* bm   = (const float*)d_in[27];
  const float* Wout = (const float*)d_in[28];
  const float* bout = (const float*)d_in[29];
  float* out = (float*)d_out;

  // --- workspace layout ---
  char* ws = (char*)d_ws;
  _Float16* W16 = (_Float16*)ws;                 // 25 matrices x 65536 halfs
  size_t off = (size_t)25 * 65536 * sizeof(_Float16);
  auto alloc = [&](size_t nfloat) -> float* {
    float* p = (float*)(ws + off);
    off += nfloat * sizeof(float);
    return p;
  };
  float* xw    = alloc((size_t)XROWS * 2048);
  float* a_s   = alloc(XROWS * 8);
  float* a_d   = alloc(XROWS * 8);
  float* x     = alloc((size_t)XROWS * Hn);
  float* e     = alloc((size_t)EROWS * Hn);      // 64 MB
  float* e_tmp = alloc((size_t)EROWS * Hn);      // 64 MB
  float* gmean = alloc(EROWS);
  float* Vx    = alloc((size_t)XROWS * Hn);
  float* Qm    = alloc((size_t)XROWS * Hn);
  float* Km    = alloc((size_t)XROWS * Hn);
  float* Vtm   = alloc((size_t)XROWS * Hn);
  float* attb  = alloc((size_t)XROWS * Hn);
  float* x_tmp = alloc((size_t)XROWS * Hn);
  // double-buffered edge BN stats (layer parity) + node stats; contiguous pairs
  float* esum0 = alloc(256);
  float* esq0  = alloc(256);
  float* esum1 = alloc(256);
  float* esq1  = alloc(256);
  float* xsum  = alloc(256);
  float* xsq   = alloc(256);

  const int MAT = 65536;  // 256*256
  // f16 weights: [Wu(0..3) Wv(4..7) Wq(8..11) Wk(12..15) Wval(16..19) Wo(20..23) Wm(24)]
  f32_to_f16<<<(4 * MAT + 255) / 256, 256, 0, stream>>>(Wu,   W16 +  0 * MAT, 4 * MAT);
  f32_to_f16<<<(4 * MAT + 255) / 256, 256, 0, stream>>>(Wv,   W16 +  4 * MAT, 4 * MAT);
  f32_to_f16<<<(4 * MAT + 255) / 256, 256, 0, stream>>>(Wq,   W16 +  8 * MAT, 4 * MAT);
  f32_to_f16<<<(4 * MAT + 255) / 256, 256, 0, stream>>>(Wk,   W16 + 12 * MAT, 4 * MAT);
  f32_to_f16<<<(4 * MAT + 255) / 256, 256, 0, stream>>>(Wval, W16 + 16 * MAT, 4 * MAT);
  f32_to_f16<<<(4 * MAT + 255) / 256, 256, 0, stream>>>(Wo,   W16 + 20 * MAT, 4 * MAT);
  f32_to_f16<<<(MAT + 255) / 256, 256, 0, stream>>>(Wm,       W16 + 24 * MAT, MAT);

  // GAT embedding + edge embedding
  gat_xw<<<XROWS, 256, 0, stream>>>(coord, W_gat, att_src, att_dst, xw, a_s, a_d);
  gat_attn<<<XROWS, 256, 0, stream>>>(x_edges, a_s, a_d, xw, b_gat, x);
  e_init<<<EROWS, 256, 0, stream>>>(x_edges, x_edges_values, W_ev, emb_edges, e);

  const float invE = 1.0f / (float)EROWS;
  for (int l = 0; l < Ln; ++l) {
    float* curS  = (l & 1) ? esum1 : esum0;   // this layer's edge stats (written)
    float* curQ  = (l & 1) ? esq1  : esq0;
    float* prevS = (l & 1) ? esum0 : esum1;   // previous layer's (read in prologue)
    float* prevQ = (l & 1) ? esq0  : esq1;

    // node-side GEMMs (M = 512)
    gemm256_wmma<<<XROWS / 64, 128, 0, stream>>>(x, W16 + (4  + l) * MAT, bv   + l * 256, Vx,  0);
    gemm256_wmma<<<XROWS / 64, 128, 0, stream>>>(x, W16 + (8  + l) * MAT, bq   + l * 256, Qm,  0);
    gemm256_wmma<<<XROWS / 64, 128, 0, stream>>>(x, W16 + (12 + l) * MAT, bk   + l * 256, Km,  0);
    gemm256_wmma<<<XROWS / 64, 128, 0, stream>>>(x, W16 + (16 + l) * MAT, bval + l * 256, Vtm, 0);

    // zero this layer's edge stats (curS/curQ contiguous) + node stats
    zero_buf<<<2, 256, 0, stream>>>(curS, 512);
    zero_buf<<<2, 256, 0, stream>>>(xsum, 512);

    // fused edge GEMM: [prologue: e += relu(BN(e_tmp, prev stats)) if l>0]
    //                  e_tmp = e@Wu + bu + Vx_i + Vx_j ; gmean ; cur stats
    edge_gemm_fused<<<EROWS / 64, 128, 0, stream>>>(
        e, (l > 0) ? e : nullptr, (l > 0) ? e_tmp : nullptr,
        prevS, prevQ, g_edge + (l - 1) * 256, be_edge + (l - 1) * 256, invE,
        W16 + (0 + l) * MAT, bu + l * 256,
        /*mode=*/0,
        Vx, e_tmp, gmean, curS, curQ,
        nullptr, nullptr, nullptr);

    // gated attention -> attb ; x_tmp = attb @ Wo + bo ; node BN stats
    node_attention<<<Bsz * NHn * Vn, 128, 0, stream>>>(Qm, Km, Vtm, gmean, attb);
    gemm256_wmma<<<XROWS / 64, 128, 0, stream>>>(attb, W16 + (20 + l) * MAT, bo + l * 256, x_tmp, 0);
    chan_stats<<<XROWS / 256, 256, 0, stream>>>(x_tmp, xsum, xsq, 256);
    residual_bn_relu<<<XROWS, 256, 0, stream>>>(x, x_tmp, xsum, xsq,
                                                g_node + l * 256, be_node + l * 256,
                                                1.0f / (float)XROWS);
  }

  // head: prologue finalizes e with layer-3 update (no write-back needed),
  // then h = relu(e@Wm + bm) reduced against Wout in-register -> out.
  edge_gemm_fused<<<EROWS / 64, 128, 0, stream>>>(
      e, nullptr, e_tmp,
      esum1, esq1, g_edge + 3 * 256, be_edge + 3 * 256, invE,
      W16 + 24 * MAT, bm,
      /*mode=*/1,
      nullptr, nullptr, nullptr, nullptr, nullptr,
      Wout, bout, out);
}
